// SEMModel_29618094474114
// MI455X (gfx1250) — matrix operand order repaired and checked by reading, not compiled, not run on hardware
//
#include <hip/hip_runtime.h>
#include <math.h>

typedef __attribute__((ext_vector_type(16))) _Float16 v16h;
typedef __attribute__((ext_vector_type(8)))  _Float16 v8h;
typedef __attribute__((ext_vector_type(8)))  float    v8f;

#define BB 4
#define SS 2048
#define DD 1024
#define MM (BB * SS)                   // 8192 rows of psi
#define NTOT ((size_t)BB * SS * DD)    // 8388608
#define NLOPS_ 4

static __device__ __forceinline__ v16h cat8(v8h lo, v8h hi) {
    return __builtin_shufflevector(lo, hi, 0, 1, 2, 3, 4, 5, 6, 7,
                                           8, 9, 10, 11, 12, 13, 14, 15);
}

#define WMMA_F16(A, B, C) \
    __builtin_amdgcn_wmma_f32_16x16x32_f16(false, (A), false, (B), (short)0, (C), false, false)

// ---------------------------------------------------------------------------
// f32 -> f16 elementwise (psi for layer 0)
// ---------------------------------------------------------------------------
__global__ __launch_bounds__(256)
void f32to16_kernel(const float* __restrict__ a, const float* __restrict__ b,
                    _Float16* __restrict__ oa, _Float16* __restrict__ ob)
{
    size_t idx = (size_t)blockIdx.x * 256 + threadIdx.x;
    oa[idx] = (_Float16)a[idx];
    ob[idx] = (_Float16)b[idx];
}

// ---------------------------------------------------------------------------
// Transpose f32 [D,D] -> f16 [D,D]^T via 32x32 LDS tile (for W)
// ---------------------------------------------------------------------------
__global__ __launch_bounds__(256)
void transpose_w_kernel(const float* __restrict__ src, _Float16* __restrict__ dst)
{
    __shared__ float tile[32][33];
    const int tilesPerRow = DD / 32;           // 32
    int tr = blockIdx.x / tilesPerRow;         // src row block
    int tc = blockIdx.x % tilesPerRow;         // src col block
    int tx = threadIdx.x & 31, ty = threadIdx.x >> 5;
#pragma unroll
    for (int r = 0; r < 4; ++r) {
        int row = tr * 32 + ty + r * 8;
        tile[ty + r * 8][tx] = src[(size_t)row * DD + tc * 32 + tx];
    }
    __syncthreads();
#pragma unroll
    for (int r = 0; r < 4; ++r) {
        int orow = tc * 32 + ty + r * 8;       // dst row = src col
        dst[(size_t)orow * DD + tr * 32 + tx] = (_Float16)tile[tx][ty + r * 8];
    }
}

// delta-store epilogue helper: writes one 16x16 subtile, returns its |delta|^2
static __device__ __forceinline__
float store_delta(v8f cr, v8f ci, int rbase, int col,
                  const float* __restrict__ psi_r, const float* __restrict__ psi_i,
                  float* __restrict__ dr, float* __restrict__ di)
{
    float s = 0.f;
#pragma unroll
    for (int v = 0; v < 8; ++v) {
        size_t idx = (size_t)(rbase + v) * DD + col;
        float er = cr[v] - psi_r[idx];
        float ei = ci[v] - psi_i[idx];
        dr[idx] = er;  di[idx] = ei;
        s += er * er + ei * ei;
    }
    return s;
}

// ---------------------------------------------------------------------------
// Complex GEMM with 2x2 register blocking: each wave computes a 32x32 output
// macro-tile (4 complex 16x16 tiles). Per K-step(32): 16 b128 loads, 16 WMMAs.
// delta = psi @ (Wr + i Wi) - psi ; per-block sum |delta|^2.
// ---------------------------------------------------------------------------
__global__ __launch_bounds__(256)
void gemm_delta_kernel(const _Float16* __restrict__ p16r, const _Float16* __restrict__ p16i,
                       const _Float16* __restrict__ WtR,  const _Float16* __restrict__ WtI,
                       const float* __restrict__ psi_r,   const float* __restrict__ psi_i,
                       float* __restrict__ dr, float* __restrict__ di,
                       float* __restrict__ partial)
{
    const int lane = threadIdx.x & 31;
    const int wave = threadIdx.x >> 5;
    const int tile = blockIdx.x * 8 + wave;
    const int TN2 = DD / 32;                   // 32 macro-tiles along N
    const int tm = tile / TN2;                 // 0..255
    const int tn = tile % TN2;

    const int l15 = lane & 15;
    const int akb = (lane >> 4) * 16;          // A frag: K-base per half-wave
    const int bkb = (lane >> 4) * 16;          // B frag: K-base per half-wave

    const _Float16* par0 = p16r + (size_t)(tm * 32 + l15) * DD + akb;
    const _Float16* par1 = par0 + (size_t)16 * DD;
    const _Float16* pai0 = p16i + (size_t)(tm * 32 + l15) * DD + akb;
    const _Float16* pai1 = pai0 + (size_t)16 * DD;
    const _Float16* pbr0 = WtR + (size_t)(tn * 32 + l15) * DD + bkb;
    const _Float16* pbr1 = pbr0 + (size_t)16 * DD;
    const _Float16* pbi0 = WtI + (size_t)(tn * 32 + l15) * DD + bkb;
    const _Float16* pbi1 = pbi0 + (size_t)16 * DD;

    v8f cr00 = {}, cr01 = {}, cr10 = {}, cr11 = {};
    v8f ci00 = {}, ci01 = {}, ci10 = {}, ci11 = {};

    for (int k0 = 0; k0 < DD; k0 += 32) {
        v16h ar0 = *(const v16h*)(par0 + k0);
        v16h ar1 = *(const v16h*)(par1 + k0);
        v16h ai0 = *(const v16h*)(pai0 + k0);
        v16h ai1 = *(const v16h*)(pai1 + k0);
        v16h an0 = -ai0, an1 = -ai1;
        v16h br0 = *(const v16h*)(pbr0 + k0);
        v16h br1 = *(const v16h*)(pbr1 + k0);
        v16h bi0 = *(const v16h*)(pbi0 + k0);
        v16h bi1 = *(const v16h*)(pbi1 + k0);
        __builtin_prefetch(par0 + k0 + 64, 0, 3);
        __builtin_prefetch(pbr0 + k0 + 64, 0, 3);

        // Cr += Ar*Wr - Ai*Wi ; Ci += Ar*Wi + Ai*Wr  (x4 subtiles)
        cr00 = WMMA_F16(ar0, br0, cr00);  cr00 = WMMA_F16(an0, bi0, cr00);
        ci00 = WMMA_F16(ar0, bi0, ci00);  ci00 = WMMA_F16(ai0, br0, ci00);
        cr01 = WMMA_F16(ar0, br1, cr01);  cr01 = WMMA_F16(an0, bi1, cr01);
        ci01 = WMMA_F16(ar0, bi1, ci01);  ci01 = WMMA_F16(ai0, br1, ci01);
        cr10 = WMMA_F16(ar1, br0, cr10);  cr10 = WMMA_F16(an1, bi0, cr10);
        ci10 = WMMA_F16(ar1, bi0, ci10);  ci10 = WMMA_F16(ai1, br0, ci10);
        cr11 = WMMA_F16(ar1, br1, cr11);  cr11 = WMMA_F16(an1, bi1, cr11);
        ci11 = WMMA_F16(ar1, bi1, ci11);  ci11 = WMMA_F16(ai1, br1, ci11);
    }

    // C/D layout per 16x16 tile: n = lane%16, m = 8*(lane/16) + vgpr
    const int r0 = tm * 32 + ((lane >> 4) << 3);
    const int c0 = tn * 32 + l15;
    float ssq = 0.f;
    ssq += store_delta(cr00, ci00, r0,      c0,      psi_r, psi_i, dr, di);
    ssq += store_delta(cr01, ci01, r0,      c0 + 16, psi_r, psi_i, dr, di);
    ssq += store_delta(cr10, ci10, r0 + 16, c0,      psi_r, psi_i, dr, di);
    ssq += store_delta(cr11, ci11, r0 + 16, c0 + 16, psi_r, psi_i, dr, di);

    for (int off = 16; off > 0; off >>= 1)
        ssq += __shfl_down(ssq, off, 32);
    __shared__ float sred[8];
    if (lane == 0) sred[wave] = ssq;
    __syncthreads();
    if (threadIdx.x == 0) {
        float s = 0.f;
        for (int w = 0; w < 8; ++w) s += sred[w];
        partial[blockIdx.x] = s;
    }
}

// Reduce block partials -> rms scalar (fixed-order, deterministic)
__global__ __launch_bounds__(256)
void rms_kernel(const float* __restrict__ partial, int n, float* __restrict__ rms_out)
{
    __shared__ float sm[256];
    float s = 0.f;
    for (int i = threadIdx.x; i < n; i += 256) s += partial[i];
    sm[threadIdx.x] = s;
    __syncthreads();
    for (int off = 128; off > 0; off >>= 1) {
        if (threadIdx.x < off) sm[threadIdx.x] += sm[threadIdx.x + off];
        __syncthreads();
    }
    if (threadIdx.x == 0) {
        float mean = sm[0] / (float)NTOT;
        float r = sqrtf(fmaxf(mean, 1e-12f));
        rms_out[0] = fmaxf(r, 1e-3f);
    }
}

// one_minus_damp[d] = 1 - GAMMA * sum_k L[k][d]^2
__global__ void damp_kernel(const float* __restrict__ L, float* __restrict__ omd)
{
    int d = blockIdx.x * blockDim.x + threadIdx.x;
    if (d < DD) {
        float s = 0.f;
        for (int k = 0; k < NLOPS_; ++k) { float v = L[k * DD + d]; s += v * v; }
        omd[d] = 1.0f - 0.01f * s;
    }
}

// ---------------------------------------------------------------------------
// mag = |delta| * |1-damp| / rms, fused with 32x32 LDS transpose:
// writes magT as f16 [B][D][S] so the Gram kernel gets contiguous fragments.
// ---------------------------------------------------------------------------
__global__ __launch_bounds__(256)
void magT_kernel(const float* __restrict__ dr, const float* __restrict__ di,
                 const float* __restrict__ omd, const float* __restrict__ rms,
                 _Float16* __restrict__ magT)
{
    __shared__ float tile[32][33];
    const int tilesD = DD / 32;                // 32
    const int tilesS = SS / 32;                // 64
    int b  = blockIdx.x / (tilesD * tilesS);
    int t  = blockIdx.x % (tilesD * tilesS);
    int ts = t / tilesD;
    int td = t % tilesD;
    int tx = threadIdx.x & 31, ty = threadIdx.x >> 5;
    float inv = 1.0f / rms[0];
    const size_t base = (size_t)b * SS * DD;
#pragma unroll
    for (int r = 0; r < 4; ++r) {
        int s = ts * 32 + ty + r * 8;
        int d = td * 32 + tx;
        size_t idx = base + (size_t)s * DD + d;
        float vr = dr[idx], vi = di[idx];
        tile[ty + r * 8][tx] = sqrtf(vr * vr + vi * vi) * fabsf(omd[d]) * inv;
    }
    __syncthreads();
    _Float16* out = magT + (size_t)b * DD * SS;
#pragma unroll
    for (int r = 0; r < 4; ++r) {
        int d = td * 32 + ty + r * 8;
        int s = ts * 32 + tx;
        out[(size_t)d * SS + s] = (_Float16)tile[tx][ty + r * 8];
    }
}

// ---------------------------------------------------------------------------
// H[b] = mag^T @ mag / S with 2x2 blocking: each wave computes 32x32 of H.
// Per K-step(32): 8 b128 loads, 4 WMMAs; all loads contiguous from magT.
// ---------------------------------------------------------------------------
__global__ __launch_bounds__(256)
void gram_kernel(const _Float16* __restrict__ magT, float* __restrict__ H)
{
    const int lane = threadIdx.x & 31;
    const int wave = threadIdx.x >> 5;
    int tile = blockIdx.x * 8 + wave;
    int b  = tile / (32 * 32);
    int t  = tile % (32 * 32);
    int ti = t / 32, tj = t % 32;              // 32x32 macro-tiles
    const _Float16* A = magT + (size_t)b * DD * SS;

    const int l15 = lane & 15;
    const int akb = (lane >> 4) * 16;
    const int bkb = (lane >> 4) * 16;
    const _Float16* pa0 = A + (size_t)(ti * 32 + l15) * SS + akb;
    const _Float16* pa1 = pa0 + (size_t)16 * SS;
    const _Float16* pb0 = A + (size_t)(tj * 32 + l15) * SS + bkb;
    const _Float16* pb1 = pb0 + (size_t)16 * SS;

    v8f c00 = {}, c01 = {}, c10 = {}, c11 = {};
    for (int s0 = 0; s0 < SS; s0 += 32) {
        v16h af0 = *(const v16h*)(pa0 + s0);
        v16h af1 = *(const v16h*)(pa1 + s0);
        v16h bf0 = *(const v16h*)(pb0 + s0);
        v16h bf1 = *(const v16h*)(pb1 + s0);
        c00 = WMMA_F16(af0, bf0, c00);
        c01 = WMMA_F16(af0, bf1, c01);
        c10 = WMMA_F16(af1, bf0, c10);
        c11 = WMMA_F16(af1, bf1, c11);
    }
    float* Hb = H + (size_t)b * DD * DD;
    const int i0 = ti * 32 + ((lane >> 4) << 3);
    const int j0 = tj * 32 + l15;
    const float invS = 1.0f / (float)SS;
#pragma unroll
    for (int v = 0; v < 8; ++v) {
        Hb[(size_t)(i0 + v)      * DD + j0]      = c00[v] * invS;
        Hb[(size_t)(i0 + v)      * DD + j0 + 16] = c01[v] * invS;
        Hb[(size_t)(i0 + v + 16) * DD + j0]      = c10[v] * invS;
        Hb[(size_t)(i0 + v + 16) * DD + j0 + 16] = c11[v] * invS;
    }
}

// Per-batch Frobenius sums: accum[b] += sum (H-Hp)^2 ; accum[4+b] += sum Hp^2
__global__ __launch_bounds__(256)
void norms_kernel(const float* __restrict__ H, const float* __restrict__ Hp,
                  float* __restrict__ accum)
{
    const int BPB = 256;
    int b  = blockIdx.x / BPB;
    int cb = blockIdx.x % BPB;
    const size_t per = (size_t)DD * DD / BPB;  // 4096
    const float* Hb  = H  + (size_t)b * DD * DD + (size_t)cb * per;
    const float* Hpb = Hp + (size_t)b * DD * DD + (size_t)cb * per;
    float sd = 0.f, sp = 0.f;
    for (size_t i = threadIdx.x; i < per; i += 256) {
        float h = Hb[i], hp = Hpb[i];
        float df = h - hp;
        sd += df * df;  sp += hp * hp;
    }
    __shared__ float smd[256], smp[256];
    smd[threadIdx.x] = sd; smp[threadIdx.x] = sp;
    __syncthreads();
    for (int off = 128; off > 0; off >>= 1) {
        if (threadIdx.x < off) {
            smd[threadIdx.x] += smd[threadIdx.x + off];
            smp[threadIdx.x] += smp[threadIdx.x + off];
        }
        __syncthreads();
    }
    if (threadIdx.x == 0) {
        atomicAdd(&accum[b],     smd[0]);
        atomicAdd(&accum[4 + b], smp[0]);
    }
}

// Diagonal condition + curvature flags -> complex per-batch scale factor
__global__ __launch_bounds__(256)
void flags_kernel(const float* __restrict__ H, const float* __restrict__ accum,
                  const float* __restrict__ theta, const float* __restrict__ jump_scale,
                  const float* __restrict__ kappa_th, int layer,
                  float* __restrict__ scale_ri)
{
    int b = blockIdx.x;
    const float* Hb = H + (size_t)b * DD * DD;
    float mx = -3.4e38f, mn = 3.4e38f;
    for (int d = threadIdx.x; d < DD; d += 256) {
        float v = Hb[(size_t)d * DD + d];
        mx = fmaxf(mx, v);  mn = fminf(mn, v);
    }
    __shared__ float smx[256], smn[256];
    smx[threadIdx.x] = mx; smn[threadIdx.x] = mn;
    __syncthreads();
    for (int off = 128; off > 0; off >>= 1) {
        if (threadIdx.x < off) {
            smx[threadIdx.x] = fmaxf(smx[threadIdx.x], smx[threadIdx.x + off]);
            smn[threadIdx.x] = fminf(smn[threadIdx.x], smn[threadIdx.x + off]);
        }
        __syncthreads();
    }
    if (threadIdx.x == 0) {
        float K = sqrtf(accum[b]) / (sqrtf(accum[4 + b]) + 1e-6f);
        bool jumped  = K > kappa_th[layer];
        bool escaped = (smx[0] / (smn[0] + 1e-12f)) > 100.0f;
        float jf = jumped ? jump_scale[layer] : 1.0f;
        float sr = jf, si = 0.f;
        if (escaped) { float th = theta[layer]; sr = jf * cosf(th); si = jf * sinf(th); }
        scale_ri[b] = sr;  scale_ri[4 + b] = si;
    }
}

// psi = residual + delta * (1-damp[d]) * scale[b]; optionally emit f16 psi
// for the next layer's GEMM A-operand (rms cancels algebraically).
__global__ __launch_bounds__(256)
void update_kernel(const float* res_r, const float* res_i,
                   const float* __restrict__ dr, const float* __restrict__ di,
                   const float* __restrict__ omd, const float* __restrict__ scale_ri,
                   float* out_r, float* out_i,
                   _Float16* o16r, _Float16* o16i, int write16)
{
    size_t idx = (size_t)blockIdx.x * 256 + threadIdx.x;
    int d = (int)(idx & (DD - 1));
    int b = (int)(idx >> 21);                  // S*D = 2^21
    float sr = scale_ri[b], si = scale_ri[4 + b];
    float f  = omd[d];
    float er = dr[idx] * f, ei = di[idx] * f;
    float vr = res_r[idx] + (er * sr - ei * si);
    float vi = res_i[idx] + (er * si + ei * sr);
    out_r[idx] = vr;
    out_i[idx] = vi;
    if (write16) { o16r[idx] = (_Float16)vr; o16i[idx] = (_Float16)vi; }
}

extern "C" void kernel_launch(void* const* d_in, const int* in_sizes, int n_in,
                              void* d_out, int out_size, void* d_ws, size_t ws_size,
                              hipStream_t stream)
{
    const float* x_real  = (const float*)d_in[0];
    const float* x_imag  = (const float*)d_in[1];
    const float* H_prev  = (const float*)d_in[2];
    const float* W_real  = (const float*)d_in[3];   // [2,D,D]
    const float* W_imag  = (const float*)d_in[4];
    const float* L_ops   = (const float*)d_in[5];   // [2,4,D]
    const float* theta   = (const float*)d_in[6];
    const float* jscale  = (const float*)d_in[7];
    const float* kth     = (const float*)d_in[8];

    float* out_r = (float*)d_out;
    float* out_i = out_r + NTOT;

    // workspace layout (byte offsets, all power-of-two sized)
    char* w = (char*)d_ws;
    float*     delta_r = (float*)w;                    w += NTOT * 4;
    float*     delta_i = (float*)w;                    w += NTOT * 4;
    _Float16*  pr16    = (_Float16*)w;                 w += NTOT * 2;
    _Float16*  pi16    = (_Float16*)w;                 w += NTOT * 2;
    _Float16*  magT    = (_Float16*)w;                 w += NTOT * 2;
    _Float16*  WtR     = (_Float16*)w;                 w += (size_t)DD * DD * 2;
    _Float16*  WtI     = (_Float16*)w;                 w += (size_t)DD * DD * 2;
    float*     H0      = (float*)w;                    w += (size_t)BB * DD * DD * 4;
    float*     H1      = (float*)w;                    w += (size_t)BB * DD * DD * 4;
    float*     partial = (float*)w;                    w += 4096 * 4;
    float*     rmsv    = (float*)w;                    w += 4;
    float*     omd     = (float*)w;                    w += DD * 4;
    float*     accum   = (float*)w;                    w += 8 * 4;
    float*     scale   = (float*)w;

    const int gemm_blocks = (MM / 32) * (DD / 32) / 8;       // 1024
    const int gram_blocks = BB * (DD / 32) * (DD / 32) / 8;  // 512
    const int ew_blocks   = (int)(NTOT / 256);               // 32768
    const int wt_blocks   = (DD / 32) * (DD / 32);           // 1024
    const int magT_blocks = BB * (DD / 32) * (SS / 32);      // 8192

    f32to16_kernel<<<ew_blocks, 256, 0, stream>>>(x_real, x_imag, pr16, pi16);

    for (int l = 0; l < 2; ++l) {
        const float* psi_r = (l == 0) ? x_real : out_r;
        const float* psi_i = (l == 0) ? x_imag : out_i;
        const float* Wr = W_real + (size_t)l * DD * DD;
        const float* Wi = W_imag + (size_t)l * DD * DD;
        const float* Hp = (l == 0) ? H_prev : H0;
        float* Hc = (l == 0) ? H0 : H1;

        hipMemsetAsync(accum, 0, 8 * sizeof(float), stream);
        transpose_w_kernel<<<wt_blocks, 256, 0, stream>>>(Wr, WtR);
        transpose_w_kernel<<<wt_blocks, 256, 0, stream>>>(Wi, WtI);

        gemm_delta_kernel<<<gemm_blocks, 256, 0, stream>>>(pr16, pi16, WtR, WtI,
                                                           psi_r, psi_i,
                                                           delta_r, delta_i, partial);
        rms_kernel<<<1, 256, 0, stream>>>(partial, gemm_blocks, rmsv);
        damp_kernel<<<4, 256, 0, stream>>>(L_ops + (size_t)l * NLOPS_ * DD, omd);
        magT_kernel<<<magT_blocks, 256, 0, stream>>>(delta_r, delta_i, omd, rmsv, magT);
        gram_kernel<<<gram_blocks, 256, 0, stream>>>(magT, Hc);
        norms_kernel<<<BB * 256, 256, 0, stream>>>(Hc, Hp, accum);
        flags_kernel<<<BB, 256, 0, stream>>>(Hc, accum, theta, jscale, kth, l, scale);
        update_kernel<<<ew_blocks, 256, 0, stream>>>(psi_r, psi_i, delta_r, delta_i,
                                                     omd, scale, out_r, out_i,
                                                     pr16, pi16, (l == 0) ? 1 : 0);
    }
}
